// Pooling_Classifier_56289841381417
// MI455X (gfx1250) — compile-verified, matrix operand order checked by reading, not run on hardware
//
#include <hip/hip_runtime.h>

// Problem constants (from the reference)
#define BSETS 4096
#define KSET  20
#define CDIM  512
#define NCLS  8631

// ---- plain ext-vector types
typedef __attribute__((ext_vector_type(4)))  float         f32x4;
typedef __attribute__((ext_vector_type(2)))  unsigned int  u32x2;
typedef __attribute__((ext_vector_type(4)))  unsigned int  u32x4;
typedef __attribute__((ext_vector_type(4)))  int           i32x4;
typedef __attribute__((ext_vector_type(16))) __bf16        v16bf;
typedef __attribute__((ext_vector_type(8)))  float         v8f;

struct Frag { u32x4 lo; u32x4 hi; };   // 32 bytes == v16bf

// address-space-qualified 128-bit vector pointers for the async-copy builtin
typedef __attribute__((address_space(1))) i32x4 gi32x4;   // global
typedef __attribute__((address_space(3))) i32x4 li32x4;   // LDS

#if defined(__has_builtin)
#if __has_builtin(__builtin_amdgcn_global_load_async_to_lds_b128)
#define HAVE_ASYNC_LDS 1
#endif
#endif

__device__ __forceinline__ unsigned int bf16rne(float f) {
    unsigned int u = __float_as_uint(f);
    u += 0x7FFFu + ((u >> 16) & 1u);          // round-to-nearest-even
    return u >> 16;
}
__device__ __forceinline__ unsigned int pack2(float a, float b) {
    return bf16rne(a) | (bf16rne(b) << 16);
}

// 16-byte global -> LDS copy: async (ASYNCcnt-tracked, bypasses VGPRs) if the
// gfx1250 builtin exists, else a plain vector copy.
__device__ __forceinline__ void g2l_b128(const unsigned short* g, unsigned short* l) {
#if HAVE_ASYNC_LDS
    __builtin_amdgcn_global_load_async_to_lds_b128(
        (gi32x4*)(unsigned long long)g,
        (li32x4*)(unsigned int)(unsigned long long)l,
        /*offset=*/0, /*cpol=*/0);
#else
    *(u32x4*)l = *(const u32x4*)g;
#endif
}

__device__ __forceinline__ void wait_async_copies() {
#if HAVE_ASYNC_LDS
#if __has_builtin(__builtin_amdgcn_s_wait_asynccnt)
    __builtin_amdgcn_s_wait_asynccnt(0);
#else
    asm volatile("s_wait_asynccnt 0x0" ::: "memory");
#endif
#endif
}

// ---------------------------------------------------------------------------
// One-shot fp32 -> bf16 conversion (for W_att / W_fc), n4 = count/4
// ---------------------------------------------------------------------------
__global__ __launch_bounds__(256)
void f32_to_bf16_kernel(const float* __restrict__ src,
                        unsigned short* __restrict__ dst, int n4)
{
    int i = blockIdx.x * 256 + threadIdx.x;
    if (i < n4) {
        f32x4 f = ((const f32x4*)src)[i];
        u32x2 p;
        p.x = pack2(f.x, f.y);
        p.y = pack2(f.z, f.w);
        ((u32x2*)dst)[i] = p;
    }
}

// ---------------------------------------------------------------------------
// Fused NAN pooling pass, one workgroup per set b:
//   xn = l2norm(x[b]); s = softmax_k(<query, xn_k>); pooled = sum_k s_k xn_k
// per_b: 0 -> query = qbase[0:C], 1 -> query = qbase[b*C:]
// Outputs: optional fp32 (d_out feats) and optional packed-bf16 (GEMM input).
// ---------------------------------------------------------------------------
__global__ __launch_bounds__(256)
void nan_pool_kernel(const float* __restrict__ x,
                     const float* __restrict__ qbase,
                     int per_b,
                     float* __restrict__ out_f32,
                     unsigned short* __restrict__ out_bf16)
{
    __shared__ float sx[KSET][CDIM];      // 40 KB raw tile (320 KB WGP LDS)
    __shared__ float sinv[KSET];
    __shared__ float sdot[KSET];
    __shared__ float scw[KSET];

    const int b    = blockIdx.x;
    const int tid  = threadIdx.x;
    const int lane = tid & 31;
    const int wv   = tid >> 5;            // 8 waves (wave32)

    const float* xb = x + (size_t)b * (KSET * CDIM);
    const float* qp = per_b ? (qbase + (size_t)b * CDIM) : qbase;

    const int NV4 = KSET * CDIM / 4;      // 2560
    for (int i = tid; i < NV4; i += 256)
        ((f32x4*)&sx[0][0])[i] = ((const f32x4*)xb)[i];
    __syncthreads();

    // per-row norm + query dot: one wave per row, shuffle reduction
    for (int r = wv; r < KSET; r += 8) {
        float sn = 0.f, sd = 0.f;
        for (int c = lane; c < CDIM; c += 32) {
            float v = sx[r][c];
            sn += v * v;
            sd += v * qp[c];
        }
        #pragma unroll
        for (int off = 16; off; off >>= 1) {
            sn += __shfl_xor(sn, off, 32);
            sd += __shfl_xor(sd, off, 32);
        }
        if (lane == 0) {
            sinv[r] = 1.0f / fmaxf(sqrtf(sn), 1e-12f);
            sdot[r] = sd;
        }
    }
    __syncthreads();

    if (tid == 0) {                        // 20-way softmax, trivial
        float w[KSET];
        float mx = -3.4e38f;
        #pragma unroll
        for (int k = 0; k < KSET; ++k) {
            w[k] = sdot[k] * sinv[k];
            mx = fmaxf(mx, w[k]);
        }
        float s = 0.f;
        #pragma unroll
        for (int k = 0; k < KSET; ++k) { w[k] = __expf(w[k] - mx); s += w[k]; }
        float is = 1.0f / s;
        #pragma unroll
        for (int k = 0; k < KSET; ++k) scw[k] = w[k] * is * sinv[k];
    }
    __syncthreads();

    // weighted sum: each thread owns a pair of adjacent channels
    const int c = tid * 2;                 // CDIM/2 == 256 pairs
    float a0 = 0.f, a1 = 0.f;
    #pragma unroll
    for (int k = 0; k < KSET; ++k) {
        a0 += scw[k] * sx[k][c];
        a1 += scw[k] * sx[k][c + 1];
    }
    const size_t o = (size_t)b * CDIM + c;
    if (out_f32) {
        out_f32[o]     = a0;
        out_f32[o + 1] = a1;
    }
    if (out_bf16)
        ((unsigned int*)out_bf16)[o >> 1] = pack2(a0, a1);
}

// ---------------------------------------------------------------------------
// WMMA bf16 GEMM: D[M][N] = op(A[M][Kd] * Bw[N][Kd]^T + bias[N]), bf16 inputs.
// 128x128 block tile, BK=32, 8 waves -> 32x64 per wave = 2x4 WMMA tiles.
// Double-buffered LDS; tiles staged with async global->LDS copies.
// ---------------------------------------------------------------------------
template <bool TANH>
__global__ __launch_bounds__(256)
void gemm_bf16_kernel(const unsigned short* __restrict__ A,
                      const unsigned short* __restrict__ Bw,
                      const float* __restrict__ bias,
                      float* __restrict__ D,
                      int M, int N, int Kd)
{
    __shared__ alignas(16) unsigned short sA[2][128][40];   // 20 KB
    __shared__ alignas(16) unsigned short sB[2][128][40];   // 20 KB

    const int tid    = threadIdx.x;
    const int lane   = tid & 31;
    const int wv     = tid >> 5;
    const int wm     = (wv & 3) * 32;
    const int wn     = (wv >> 2) * 64;
    const int mrow   = lane & 15;
    const int half   = lane >> 4;
    const int blockM = blockIdx.y * 128;
    const int blockN = blockIdx.x * 128;

    v8f acc[2][4];
    const v8f vzero = {0.f, 0.f, 0.f, 0.f, 0.f, 0.f, 0.f, 0.f};
    #pragma unroll
    for (int i = 0; i < 2; ++i)
        #pragma unroll
        for (int j = 0; j < 4; ++j) acc[i][j] = vzero;

    // stage one 128x32-bf16 tile of A and B into LDS buffer `buf`
    auto stage = [&](int buf, int k0) {
        #pragma unroll
        for (int s = 0; s < 2; ++s) {
            const int cchunk = tid + s * 256;   // 512 x 16B chunks per matrix
            const int row = cchunk >> 2;
            const int ch  = cchunk & 3;
            const unsigned short* ga =
                A + (size_t)(blockM + row) * Kd + k0 + ch * 8;
            g2l_b128(ga, &sA[buf][row][ch * 8]);
            int gn = blockN + row;
            if (gn > N - 1) gn = N - 1;         // clamp: masked at store anyway
            const unsigned short* gb =
                Bw + (size_t)gn * Kd + k0 + ch * 8;
            g2l_b128(gb, &sB[buf][row][ch * 8]);
        }
    };

    stage(0, 0);
    wait_async_copies();
    __syncthreads();

    const int nk = Kd >> 5;
    for (int t = 0; t < nk; ++t) {
        const int cur = t & 1;
        if (t + 1 < nk) stage(cur ^ 1, (t + 1) << 5);

        // fragment gather (ISA 16-bit A layout; B mirrored N<->M)
        Frag fa2[2], fb2[4];
        #pragma unroll
        for (int i = 0; i < 2; ++i) {
            const int r = wm + i * 16 + mrow;
            fa2[i].lo = *(const u32x4*)&sA[cur][r][half * 8];
            fa2[i].hi = *(const u32x4*)&sA[cur][r][16 + half * 8];
        }
        #pragma unroll
        for (int j = 0; j < 4; ++j) {
            const int r = wn + j * 16 + mrow;
            fb2[j].lo = *(const u32x4*)&sB[cur][r][half * 8];
            fb2[j].hi = *(const u32x4*)&sB[cur][r][16 + half * 8];
        }

        #pragma unroll
        for (int i = 0; i < 2; ++i) {
            v16bf av = __builtin_bit_cast(v16bf, fa2[i]);
            #pragma unroll
            for (int j = 0; j < 4; ++j) {
                v16bf bv = __builtin_bit_cast(v16bf, fb2[j]);
                acc[i][j] = __builtin_amdgcn_wmma_f32_16x16x32_bf16(
                    false, av, false, bv, (short)0, acc[i][j], false, false);
            }
        }

        wait_async_copies();
        __syncthreads();
    }

    // epilogue: C/D layout -> col = lane&15, rows rr + 8*half
    #pragma unroll
    for (int i = 0; i < 2; ++i) {
        #pragma unroll
        for (int j = 0; j < 4; ++j) {
            const int col = blockN + wn + j * 16 + mrow;
            if (col < N) {
                const float bv = bias[col];
                #pragma unroll
                for (int rr = 0; rr < 8; ++rr) {
                    const int row = blockM + wm + i * 16 + half * 8 + rr;
                    float v = acc[i][j][rr] + bv;
                    if (TANH) v = tanhf(v);
                    D[(size_t)row * N + col] = v;
                }
            }
        }
    }
}

// ---------------------------------------------------------------------------
// Launch pipeline. d_out = [feats fp32 | logits fp32].
// ws: r_bf16 | feats_bf16 | new_q fp32 | W_att_bf16 | W_fc_bf16  (~25 MB)
// ---------------------------------------------------------------------------
extern "C" void kernel_launch(void* const* d_in, const int* in_sizes, int n_in,
                              void* d_out, int out_size, void* d_ws, size_t ws_size,
                              hipStream_t stream)
{
    (void)in_sizes; (void)n_in; (void)out_size; (void)ws_size;

    const float* x     = (const float*)d_in[0];
    // d_in[1] = lst_lens (uniform K, unused)
    const float* q     = (const float*)d_in[2];
    const float* W_att = (const float*)d_in[3];
    const float* b_att = (const float*)d_in[4];
    const float* W_fc  = (const float*)d_in[5];
    const float* b_fc  = (const float*)d_in[6];

    float* feats  = (float*)d_out;
    float* logits = feats + (size_t)BSETS * CDIM;

    char* ws = (char*)d_ws;
    unsigned short* r_bf16     = (unsigned short*)ws;                    // 4 MB
    unsigned short* feats_bf16 = r_bf16 + (size_t)BSETS * CDIM;          // 4 MB
    float*          new_q      = (float*)(feats_bf16 + (size_t)BSETS * CDIM); // 8 MB
    unsigned short* Watt_bf16  = (unsigned short*)(new_q + (size_t)BSETS * CDIM);
    unsigned short* Wfc_bf16   = Watt_bf16 + (size_t)CDIM * CDIM;

    // one-shot weight conversions to bf16
    {
        int n4 = (CDIM * CDIM) / 4;
        f32_to_bf16_kernel<<<(n4 + 255) / 256, 256, 0, stream>>>(W_att, Watt_bf16, n4);
    }
    {
        int n4 = (NCLS * CDIM) / 4;
        f32_to_bf16_kernel<<<(n4 + 255) / 256, 256, 0, stream>>>(W_fc, Wfc_bf16, n4);
    }

    // pass 1: global query -> r (bf16 only; feeds tanh-GEMM)
    nan_pool_kernel<<<BSETS, 256, 0, stream>>>(x, q, 0, nullptr, r_bf16);

    // new_q = tanh(r @ W_att^T + b_att)   (fp32 out; consumed as query)
    dim3 g2((CDIM + 127) / 128, BSETS / 128);
    gemm_bf16_kernel<true><<<g2, 256, 0, stream>>>(r_bf16, Watt_bf16, b_att,
                                                   new_q, BSETS, CDIM, CDIM);

    // pass 2: adapted query -> feats (fp32 to d_out + bf16 for classifier)
    nan_pool_kernel<<<BSETS, 256, 0, stream>>>(x, new_q, 1, feats, feats_bf16);

    // logits = feats @ W_fc^T + b_fc
    dim3 g4((NCLS + 127) / 128, BSETS / 128);
    gemm_bf16_kernel<false><<<g4, 256, 0, stream>>>(feats_bf16, Wfc_bf16, b_fc,
                                                    logits, BSETS, NCLS, CDIM);
}
